// Sim_71683004170886
// MI455X (gfx1250) — matrix operand order repaired and checked by reading, not compiled, not run on hardware
//
#include <hip/hip_runtime.h>

typedef _Float16 f16;
typedef __attribute__((ext_vector_type(16))) _Float16 v16h;
typedef __attribute__((ext_vector_type(8)))  _Float16 v8h;
typedef __attribute__((ext_vector_type(8)))  float    v8f;

#define NCH 48
#define CH_STRIDE 262144          // 64*64*64
#define WROW 64                   // padded K per window row (f16)
#define WSTRIDE 528               // 8*64 + 16 pad (f16) -> 1056 B, 32B aligned per window

__global__ __launch_bounds__(256) void swin_aff_kernel(
    const float* __restrict__ x_in, const float* __restrict__ y_in,
    const float* __restrict__ nxw, const float* __restrict__ nxb,
    const float* __restrict__ nyw, const float* __restrict__ nyb,
    float* __restrict__ out)
{
    __shared__ __align__(32) f16 xh[32 * WSTRIDE];
    __shared__ __align__(32) f16 yh[32 * WSTRIDE];
    __shared__ float asq_lds[256];       // [window][i]
    __shared__ float outbuf[3 * 256];    // [t][r][w]
    __shared__ float lnw[4 * NCH];       // nxw | nxb | nyw | nyb

    const int tid = threadIdx.x;
    const int hp = blockIdx.x, dp = blockIdx.y, bb = blockIdx.z;
    const int d0 = dp * 2, h0 = hp * 2;
    const int r  = tid >> 6, w = tid & 63;   // 4 rows x 64 voxels
    const int dd = r >> 1, hh = r & 1;
    const int spat = (d0 + dd) * 4096 + (h0 + hh) * 64 + w;
    const int wi = w >> 1;                   // window index along w (0..31)
    const int ii = dd * 4 + hh * 2 + (w & 1); // in-window voxel index (NW)

    if (tid < 4 * NCH) {
        int a = tid / NCH, c = tid - a * NCH;
        const float* src = (a == 0) ? nxw : (a == 1) ? nxb : (a == 2) ? nyw : nyb;
        lnw[tid] = src[c];
    }

    // ---------------- tensor X: layernorm -> f16 LDS + |x|^2 ----------------
    {
        const float* p = x_in + (size_t)bb * NCH * CH_STRIDE + spat;
        float v[NCH]; float s = 0.f, s2 = 0.f;
        #pragma unroll
        for (int c = 0; c < NCH; ++c) {
            float t = p[(size_t)c * CH_STRIDE];
            v[c] = t; s += t; s2 += t * t;
        }
        float mu  = s  * (1.f / NCH);
        float var = s2 * (1.f / NCH) - mu * mu;
        float rs  = rsqrtf(var + 1e-5f);
        __syncthreads();                     // LN weights visible
        f16* row = &xh[wi * WSTRIDE + ii * WROW];
        float aq = 0.f;
        #pragma unroll
        for (int c = 0; c < NCH; ++c) {
            float xn = (v[c] - mu) * rs * lnw[c] + lnw[NCH + c];
            row[c] = (f16)xn;
            aq += xn * xn;
        }
        #pragma unroll
        for (int c = NCH; c < 64; ++c) row[c] = (f16)0.f;   // zero K-pad
        asq_lds[wi * 8 + ii] = aq;
    }
    // ---------------- tensor Y ----------------
    {
        const float* p = y_in + (size_t)bb * NCH * CH_STRIDE + spat;
        float v[NCH]; float s = 0.f, s2 = 0.f;
        #pragma unroll
        for (int c = 0; c < NCH; ++c) {
            float t = p[(size_t)c * CH_STRIDE];
            v[c] = t; s += t; s2 += t * t;
        }
        float mu  = s  * (1.f / NCH);
        float var = s2 * (1.f / NCH) - mu * mu;
        float rs  = rsqrtf(var + 1e-5f);
        f16* row = &yh[wi * WSTRIDE + ii * WROW];
        #pragma unroll
        for (int c = 0; c < NCH; ++c) {
            float yn = (v[c] - mu) * rs * lnw[2 * NCH + c] + lnw[3 * NCH + c];
            row[c] = (f16)yn;
        }
        #pragma unroll
        for (int c = NCH; c < 64; ++c) row[c] = (f16)0.f;
    }
    __syncthreads();

    // ---------------- WMMA: 2 windows per 16x16 tile, K=48 (padded 64) -------
    const int lane = tid & 31, wv = tid >> 5;
    const int M  = lane & 15;        // row (A) == col (B) numerically per lane
    const int hi = lane >> 4;        // upper half-wave flag
    #pragma unroll
    for (int pp = 0; pp < 2; ++pp) {
        const int pair = wv * 2 + pp;
        const int winA = pair * 2, winB = pair * 2 + 1;
        const int rcWin = (M < 8) ? winA : winB;
        const int rc    = M & 7;

        // A fragment (contiguous order): K = hi*16 + t
        const v8h* arow = (const v8h*)&xh[rcWin * WSTRIDE + rc * WROW];
        union { v16h v; v8h h[2]; } a0, a1;
        a0.h[0] = arow[2 * hi];       a0.h[1] = arow[2 * hi + 1];   // K 0..31
        a1.h[0] = arow[4 + 2 * hi];   a1.h[1] = arow[5 + 2 * hi];   // K 32..63

        // B fragment (contiguous order, same as A): lanes 0-15 K=0..15, lanes 16-31 K=16..31
        const f16* bp = &yh[rcWin * WSTRIDE + rc * WROW];
        v16h b0 = *(const v16h*)(bp + hi * 16);             // K 0..31
        v16h b1 = *(const v16h*)(bp + 32 + hi * 16);        // K 32..63

        v8f acc = {0.f, 0.f, 0.f, 0.f, 0.f, 0.f, 0.f, 0.f};
        acc = __builtin_amdgcn_wmma_f32_16x16x32_f16(false, a0.v, false, b0,
                                                     (short)0, acc, false, false);
        acc = __builtin_amdgcn_wmma_f32_16x16x32_f16(false, a1.v, false, b1,
                                                     (short)0, acc, false, false);

        // Useful columns: lanes 0-7 (winA, rows M=0..7) and 24-31 (winB, rows M=8..15)
        if (lane < 8 || lane >= 24) {
            const int winE = (lane < 8) ? winA : winB;
            const int j = lane & 7;
            float aff[8], mx = -1e30f;
            #pragma unroll
            for (int i = 0; i < 8; ++i) {
                aff[i] = (2.f * acc[i] - asq_lds[winE * 8 + i]) * 0.14433756729740643f;
                mx = fmaxf(mx, aff[i]);
            }
            float e[8], sum = 0.f;
            #pragma unroll
            for (int i = 0; i < 8; ++i) { e[i] = __expf(aff[i] - mx); sum += e[i]; }
            const float inv = 1.f / sum;
            float sd = 0.f, sh = 0.f, sw = 0.f;
            #pragma unroll
            for (int i = 0; i < 8; ++i) {
                float pr = e[i] * inv;
                if (i & 4) sd += pr;
                if (i & 2) sh += pr;
                if (i & 1) sw += pr;
            }
            const int vr = (j >> 2) * 2 + ((j >> 1) & 1);  // row (dd*2+hh)
            const int vw = winE * 2 + (j & 1);             // w within tile
            outbuf[0 * 256 + vr * 64 + vw] = sd;
            outbuf[1 * 256 + vr * 64 + vw] = sh;
            outbuf[2 * 256 + vr * 64 + vw] = sw;
        }
    }
    __syncthreads();

    // ---------------- coalesced store: (b,3,d,h,w) ----------------
    const int dg = d0 + (r >> 1), hg = h0 + (r & 1);
    #pragma unroll
    for (int t = 0; t < 3; ++t) {
        out[(((size_t)(bb * 3 + t) * 64 + dg) * 64 + hg) * 64 + w] =
            outbuf[t * 256 + tid];
    }
}

extern "C" void kernel_launch(void* const* d_in, const int* in_sizes, int n_in,
                              void* d_out, int out_size, void* d_ws, size_t ws_size,
                              hipStream_t stream) {
    const float* x_in = (const float*)d_in[0];
    const float* y_in = (const float*)d_in[1];
    const float* nxw  = (const float*)d_in[2];
    const float* nxb  = (const float*)d_in[3];
    const float* nyw  = (const float*)d_in[4];
    const float* nyb  = (const float*)d_in[5];
    float* out = (float*)d_out;

    dim3 grid(32, 32, 2);   // (h-pairs, d-pairs, batch)
    dim3 block(256);
    swin_aff_kernel<<<grid, block, 0, stream>>>(x_in, y_in, nxw, nxb, nyw, nyb, out);
}
